// SelfOrganizingMap_79877801771115
// MI455X (gfx1250) — compile-verified, matrix operand order checked by reading, not run on hardware
//
#include <hip/hip_runtime.h>
#include <hip/hip_bf16.h>

// ---------------------------------------------------------------------------
// SOM scan, restructured for MI455X (gfx1250, wave32, WMMA):
//   Phase 1 (parallel, WMMA bf16 split-precision, software-pipelined):
//     dot0 = W0 . A^T   [4096 x 2048]   (v_wmma_f32_16x16x32_bf16, hi/lo split)
//     S    = A  . A^T   [2048 x 2048]
//     n2W[g] = ||W0[g]||^2 (exact f32),  n2A[b] = ||act[b]||^2
//   Phase 2 (sequential scan, single 1024-thread workgroup, L2/LDS-resident):
//     d2[g] = n2W[g] + n2A[b] - 2*dot[g,b]  -> global + class-constrained argmin
//     wave32 shfl_xor butterfly argmin (2 barriers/step)
//     err row output from exact f32 weights
//     rank-1 incremental maintenance of dot columns (>b), W_cur, n2W(in LDS)
// ---------------------------------------------------------------------------

#define B_N    2048
#define DIM_D  2048
#define GRID_G 64
#define NC     4096            // GRID_G * GRID_G
#define REL_THRESH_F 0.95f
#define SOM_LR_MAX_F 0.2f
#define SOM_LR_MIN_F 0.05f
#define CTX_MAX_I 2
#define CTX_MIN_I 0

typedef __attribute__((ext_vector_type(16))) __bf16 v16bf;
typedef __attribute__((ext_vector_type(8)))  __bf16 v8bf;
typedef __attribute__((ext_vector_type(8)))  float  v8f;

struct SomParams { float lr; int ctx; };

// ---------------------------------------------------------------- init params
__global__ void som_init_params(const int* __restrict__ epoch,
                                const int* __restrict__ max_epochs,
                                SomParams* __restrict__ prm) {
  if (blockIdx.x == 0 && threadIdx.x == 0) {
    int e = epoch[0], me = max_epochs[0];
    float prog = (me > 0) ? (float)(me - e) / (float)me : 0.0f;
    prog = fminf(1.0f, fmaxf(0.0f, prog));
    float p2 = prog * prog, p4 = p2 * p2;
    prm->ctx = CTX_MIN_I + (int)(p4 * (float)(CTX_MAX_I - CTX_MIN_I));
    prm->lr  = SOM_LR_MIN_F + prog * (SOM_LR_MAX_F - SOM_LR_MIN_F);
  }
}

// ------------------------------------------------- f32 -> (hi,lo) bf16 split
__global__ void som_split_bf16(const float* __restrict__ src,
                               __bf16* __restrict__ hi, __bf16* __restrict__ lo,
                               float* __restrict__ copy, int n) {
  int stride = gridDim.x * blockDim.x;
  for (int i = blockIdx.x * blockDim.x + threadIdx.x; i < n; i += stride) {
    float x = src[i];
    __bf16 h = (__bf16)x;
    hi[i] = h;
    lo[i] = (__bf16)(x - (float)h);
    if (copy) copy[i] = x;
  }
}

// ------------------------------------------------------------ exact row norms
__global__ void som_row_norm2(const float* __restrict__ src,
                              float* __restrict__ out, int cols) {
  __shared__ float s[256];
  const float* row = src + (size_t)blockIdx.x * cols;
  float acc = 0.0f;
  for (int c = threadIdx.x; c < cols; c += blockDim.x) { float v = row[c]; acc += v * v; }
  s[threadIdx.x] = acc; __syncthreads();
  for (int t = 128; t > 0; t >>= 1) {
    if ((int)threadIdx.x < t) s[threadIdx.x] += s[threadIdx.x + t];
    __syncthreads();
  }
  if (threadIdx.x == 0) out[blockIdx.x] = s[0];
}

// ---------------------------------------------------------------------------
// C[M,N] = Xrows . Yrows^T  (NT GEMM), bf16 2-term split, f32 accumulate.
// One wave32 computes a 16(M) x 64(N) tile. Three INDEPENDENT accumulators per
// N-tile (hi*hi, hi*lo, lo*hi) -> no intra-step WMMA RAW chains.
// K-loop is software-pipelined with 2 full operand buffers (A hi/lo + 4x B
// hi/lo each): while the 12-WMMA burst for K-step k executes, the loads and
// address VALU for K-step k+64 fill the WMMA->VMEM WAR co-exec slots, so the
// compiler needs no v_nop padding and loads stay >1 step ahead of their waits.
// Requires K % 64 == 0 (true here: K = 2048).
// Operand VGPR layout per ISA 7.12.2 (16-bit A-matrix 16x32):
//   lanes 0-15  : row = lane,    K in {k0..k0+7}  and {k0+16..k0+23}
//   lanes 16-31 : row = lane-16, K in {k0+8..k0+15} and {k0+24..k0+31}
// ---------------------------------------------------------------------------
__global__ __launch_bounds__(256)
void som_gemm_nt_split(const __bf16* __restrict__ Xhi, const __bf16* __restrict__ Xlo,
                       const __bf16* __restrict__ Yhi, const __bf16* __restrict__ Ylo,
                       float* __restrict__ C, int M, int N, int K) {
  const int wid  = (blockIdx.x * blockDim.x + threadIdx.x) >> 5;
  const int lane = threadIdx.x & 31;
  const int tilesM = M >> 4;
  const int tilesN = N >> 6;
  const int tm = wid % tilesM;
  const int tn = wid / tilesM;
  if (tn >= tilesN) return;

  const int half = lane >> 4;          // 0: lanes 0-15, 1: lanes 16-31
  const int mr   = lane & 15;
  const int koff = half ? 8 : 0;

  const __bf16* xh = Xhi + (size_t)(tm * 16 + mr) * K;
  const __bf16* xl = Xlo + (size_t)(tm * 16 + mr) * K;
  const __bf16* yh[4]; const __bf16* yl[4];
#pragma unroll
  for (int j = 0; j < 4; ++j) {
    int nrow = tn * 64 + j * 16 + mr;
    yh[j] = Yhi + (size_t)nrow * K;
    yl[j] = Ylo + (size_t)nrow * K;
  }

  v8f acc_hh[4] = {}, acc_hl[4] = {}, acc_lh[4] = {};
  union V16 { v16bf v; v8bf h[2]; };

  auto loadK = [&](int k0, V16& ah, V16& al, V16 (&bh)[4], V16 (&bl)[4]) {
    __builtin_prefetch(xh + k0 + 256, 0, 1);       // global_prefetch_b8
#pragma unroll
    for (int j = 0; j < 4; ++j) {
      bh[j].h[0] = *(const v8bf*)(yh[j] + k0 + koff);
      bh[j].h[1] = *(const v8bf*)(yh[j] + k0 + 16 + koff);
      bl[j].h[0] = *(const v8bf*)(yl[j] + k0 + koff);
      bl[j].h[1] = *(const v8bf*)(yl[j] + k0 + 16 + koff);
      __builtin_prefetch(yh[j] + k0 + 256, 0, 1);
    }
    ah.h[0] = *(const v8bf*)(xh + k0 + koff);
    ah.h[1] = *(const v8bf*)(xh + k0 + 16 + koff);
    al.h[0] = *(const v8bf*)(xl + k0 + koff);
    al.h[1] = *(const v8bf*)(xl + k0 + 16 + koff);
  };
  auto computeK = [&](V16& ah, V16& al, V16 (&bh)[4], V16 (&bl)[4]) {
#pragma unroll
    for (int j = 0; j < 4; ++j) {
      acc_hh[j] = __builtin_amdgcn_wmma_f32_16x16x32_bf16(false, ah.v, false, bh[j].v,
                                                          (short)0, acc_hh[j], false, false);
      acc_hl[j] = __builtin_amdgcn_wmma_f32_16x16x32_bf16(false, ah.v, false, bl[j].v,
                                                          (short)0, acc_hl[j], false, false);
      acc_lh[j] = __builtin_amdgcn_wmma_f32_16x16x32_bf16(false, al.v, false, bh[j].v,
                                                          (short)0, acc_lh[j], false, false);
    }
  };

  V16 a0h, a0l, b0h[4], b0l[4];
  V16 a1h, a1l, b1h[4], b1l[4];
  loadK(0,  a0h, a0l, b0h, b0l);
  loadK(32, a1h, a1l, b1h, b1l);

  for (int k0 = 0; k0 < K; k0 += 64) {
    computeK(a0h, a0l, b0h, b0l);
    if (k0 + 64 < K) loadK(k0 + 64, a0h, a0l, b0h, b0l);
    computeK(a1h, a1l, b1h, b1l);
    if (k0 + 96 < K) loadK(k0 + 96, a1h, a1l, b1h, b1l);
  }

  // C/D layout: VGPR r -> M = r + 8*half, N = lane&15
#pragma unroll
  for (int j = 0; j < 4; ++j) {
    v8f acc = acc_hh[j] + acc_hl[j] + acc_lh[j];
    const int n0 = tn * 64 + j * 16 + mr;
#pragma unroll
    for (int r = 0; r < 8; ++r) {
      int m = tm * 16 + r + (half ? 8 : 0);
      C[(size_t)m * N + n0] = acc[r];
    }
  }
}

// ---------------------------------------------------------------------------
// Sequential scan: one persistent workgroup (1024 threads = 32 wave32s).
// Per-step cost ~115K FLOP, operands L2/LDS-resident. Argmins use wave32
// shfl_xor butterflies + one 32-entry cross-wave combine (2 barriers/step).
// ---------------------------------------------------------------------------
__global__ __launch_bounds__(1024)
void som_scan(const float* __restrict__ act, const int* __restrict__ labels,
              const int* __restrict__ cell_labels, const float* __restrict__ cell_rel,
              const SomParams* __restrict__ prm,
              float* __restrict__ dot, const float* __restrict__ S,
              const float* __restrict__ n2W, const float* __restrict__ n2A,
              float* __restrict__ Wcur, float* __restrict__ out) {
  __shared__ float s_n2w[NC];            // 16 KB, live for whole scan
  __shared__ unsigned char s_clab[NC];   // 4 KB  (labels are 0..9)
  __shared__ float s_wv[32];  __shared__ int s_wi[32];
  __shared__ float s_wpv[32]; __shared__ int s_wpi[32];
  __shared__ int   s_cells[25]; __shared__ float s_pw[25];
  __shared__ int   s_nc; __shared__ int s_pidx; __shared__ float s_rel;
  __shared__ float s_lr; __shared__ int s_ctx;

  const int tid  = threadIdx.x;
  const int lane = tid & 31;
  const int wave = tid >> 5;

  if (tid == 0) { s_lr = prm->lr; s_ctx = prm->ctx; }
  for (int g = tid; g < NC; g += 1024) {
    s_n2w[g]  = n2W[g];
    s_clab[g] = (unsigned char)cell_labels[g];
  }
  __syncthreads();
  const float lr = s_lr; const int ctx = s_ctx;

  for (int b = 0; b < B_N; ++b) {
    const int   lab = labels[b];
    const float n2a = n2A[b];

    // --- distances + local argmins (global + class-constrained) ---
    float best  = __builtin_inff(); int bidx = 0;
    float pbest = __builtin_inff(); int pidx = 0;
    for (int g = tid; g < NC; g += 1024) {
      float d2 = s_n2w[g] + n2a - 2.0f * dot[(size_t)g * B_N + b];
      if (d2 < best) { best = d2; bidx = g; }
      if ((int)s_clab[g] == lab && d2 < pbest) { pbest = d2; pidx = g; }
    }
    // wave32 butterfly (symmetric combine -> all lanes converge)
#pragma unroll
    for (int m = 16; m > 0; m >>= 1) {
      float ov = __shfl_xor(best, m, 32);  int oi = __shfl_xor(bidx, m, 32);
      if (ov < best || (ov == best && oi < bidx)) { best = ov; bidx = oi; }
      float pv = __shfl_xor(pbest, m, 32); int pi = __shfl_xor(pidx, m, 32);
      if (pv < pbest || (pv == pbest && pi < pidx)) { pbest = pv; pidx = pi; }
    }
    if (lane == 0) { s_wv[wave] = best; s_wi[wave] = bidx; s_wpv[wave] = pbest; s_wpi[wave] = pidx; }
    __syncthreads();

    if (wave == 0) {
      float v  = s_wv[lane];  int i0 = s_wi[lane];
      float pv = s_wpv[lane]; int p0 = s_wpi[lane];
#pragma unroll
      for (int m = 16; m > 0; m >>= 1) {
        float ov = __shfl_xor(v, m, 32);  int oi = __shfl_xor(i0, m, 32);
        if (ov < v || (ov == v && oi < i0)) { v = ov; i0 = oi; }
        float opv = __shfl_xor(pv, m, 32); int opi = __shfl_xor(p0, m, 32);
        if (opv < pv || (opv == pv && opi < p0)) { pv = opv; p0 = opi; }
      }
      if (lane == 0) {
        const int bx = i0 >> 6, by = i0 & (GRID_G - 1);
        s_pidx = p0;
        s_rel  = cell_rel[p0] * 0.01f;
        int nc = 0;
        if (v > 1e-4f) {                         // gate on min_dist
          for (int dx = -ctx; dx <= ctx; ++dx) {
            int x = bx + dx; if (x < 0 || x >= GRID_G) continue;
            for (int dy = -ctx; dy <= ctx; ++dy) {
              int y = by + dy; if (y < 0 || y >= GRID_G) continue;
              int adx = dx < 0 ? -dx : dx, ady = dy < 0 ? -dy : dy;
              int cheb = adx > ady ? adx : ady;
              s_cells[nc] = x * GRID_G + y;
              s_pw[nc]    = lr * exp2f(-(float)cheb);   // lr * 0.5^cheb
              ++nc;
            }
          }
        }
        s_nc = nc;
      }
    }
    __syncthreads();

    // --- error output row (must use PRE-update weights) ---
    const int   pidx0 = s_pidx;
    const float rel   = s_rel;
    const bool  doErr = rel >= REL_THRESH_F;
    const float* proto = Wcur + (size_t)pidx0 * DIM_D;
    const float* a     = act  + (size_t)b     * DIM_D;
    float*       o     = out  + (size_t)b     * DIM_D;
    for (int d = tid; d < DIM_D; d += 1024)
      o[d] = doErr ? 0.01f * rel * (proto[d] - a[d]) : 0.0f;
    __syncthreads();   // err read of Wcur completes before update

    // --- neighborhood update: W_cur, dot columns (> b), ||W||^2 ---
    const int nc = s_nc;
    if (nc > 0) {
      for (int i = tid; i < nc * DIM_D; i += 1024) {
        int c = i >> 11; int d = i & (DIM_D - 1);
        int g = s_cells[c]; float p = s_pw[c];
        float w = Wcur[(size_t)g * DIM_D + d];
        Wcur[(size_t)g * DIM_D + d] = w + p * (a[d] - w);
      }
      const float* srow = S + (size_t)b * B_N;
      for (int i = tid; i < nc * B_N; i += 1024) {
        int c = i >> 11; int bp = i & (B_N - 1);
        if (bp > b) {
          int g = s_cells[c]; float p = s_pw[c];
          float dv = dot[(size_t)g * B_N + bp];
          dot[(size_t)g * B_N + bp] = dv + p * (srow[bp] - dv);
        }
      }
      if (tid < nc) {   // dot[g,b] still holds the pre-update value
        int g = s_cells[tid]; float p = s_pw[tid];
        float dwa = dot[(size_t)g * B_N + b];
        float q = 1.0f - p;
        s_n2w[g] = q * q * s_n2w[g] + 2.0f * p * q * dwa + p * p * n2a;
      }
    }
    __syncthreads();
  }
}

// ---------------------------------------------------------------------------
extern "C" void kernel_launch(void* const* d_in, const int* in_sizes, int n_in,
                              void* d_out, int out_size, void* d_ws, size_t ws_size,
                              hipStream_t stream) {
  const float* act    = (const float*)d_in[0];
  const int*   labels = (const int*)  d_in[1];
  const float* somv   = (const float*)d_in[2];
  const int*   clab   = (const int*)  d_in[3];
  const float* crel   = (const float*)d_in[4];
  const int*   epoch  = (const int*)  d_in[5];
  const int*   maxep  = (const int*)  d_in[6];
  float* out = (float*)d_out;

  char* ws = (char*)d_ws;
  size_t off = 0;
  auto carve = [&](size_t bytes) -> void* {
    void* p = ws + off;
    off += (bytes + 255) & ~(size_t)255;
    return p;
  };
  SomParams* prm = (SomParams*)carve(sizeof(SomParams));
  __bf16* Whi = (__bf16*)carve((size_t)NC  * DIM_D * sizeof(__bf16));
  __bf16* Wlo = (__bf16*)carve((size_t)NC  * DIM_D * sizeof(__bf16));
  __bf16* Ahi = (__bf16*)carve((size_t)B_N * DIM_D * sizeof(__bf16));
  __bf16* Alo = (__bf16*)carve((size_t)B_N * DIM_D * sizeof(__bf16));
  float*  dot = (float*) carve((size_t)NC  * B_N   * sizeof(float));
  float*  S   = (float*) carve((size_t)B_N * B_N   * sizeof(float));
  float*  n2W = (float*) carve((size_t)NC          * sizeof(float));
  float*  n2A = (float*) carve((size_t)B_N         * sizeof(float));
  float*  Wcur= (float*) carve((size_t)NC  * DIM_D * sizeof(float));
  (void)ws_size; (void)in_sizes; (void)n_in; (void)out_size;

  som_init_params<<<1, 1, 0, stream>>>(epoch, maxep, prm);
  som_split_bf16<<<4096, 256, 0, stream>>>(somv, Whi, Wlo, Wcur, NC * DIM_D);
  som_split_bf16<<<4096, 256, 0, stream>>>(act,  Ahi, Alo, nullptr, B_N * DIM_D);
  som_row_norm2<<<NC,  256, 0, stream>>>(somv, n2W, DIM_D);
  som_row_norm2<<<B_N, 256, 0, stream>>>(act,  n2A, DIM_D);

  { // dot = W0 . A^T   [4096 x 2048]
    int waves = (NC / 16) * (B_N / 64);          // 8192 waves
    som_gemm_nt_split<<<waves / 8, 256, 0, stream>>>(Whi, Wlo, Ahi, Alo, dot, NC, B_N, DIM_D);
  }
  { // S = A . A^T      [2048 x 2048]
    int waves = (B_N / 16) * (B_N / 64);         // 4096 waves
    som_gemm_nt_split<<<waves / 8, 256, 0, stream>>>(Ahi, Alo, Ahi, Alo, S, B_N, B_N, DIM_D);
  }
  som_scan<<<1, 1024, 0, stream>>>(act, labels, clab, crel, prm,
                                   dot, S, n2W, n2A, Wcur, out);
}
